// AutoregressiveGMM_70085276336347
// MI455X (gfx1250) — compile-verified, matrix operand order checked by reading, not run on hardware
//
#include <hip/hip_runtime.h>
#include <hip/hip_bf16.h>
#include <math.h>

// ---------------------------------------------------------------------------
// AutoregressiveGMM for MI455X (gfx1250): f16 WMMA (f32 accum) throughout.
//   base[b,h]   = context @ W0[2D:,:]          (once, WMMA)
//   step i:     h = relu(base + masktab[i])    (masktab = b0 + prefix(W0[D:2D]))
//               2x residual blocks via v_wmma_f32_16x16x32_f16
//               p = h @ Wh_r[:,i,:] (+bh)      (WMMA, N padded 30->32)
//               per-row GMM logsumexp (VALU), acc += lp
//               base += value[:,i] (x) W0[i,:] (rank-1 update in regs)
// ---------------------------------------------------------------------------

typedef _Float16 h16;
typedef __attribute__((ext_vector_type(16))) _Float16 v16h;
typedef __attribute__((ext_vector_type(8)))  _Float16 v8h;
typedef __attribute__((ext_vector_type(8)))  float    v8f;

#define NB      16384
#define DD      64
#define KK      10
#define HH      256
#define CC      512
#define ROWS    64          // batch rows per workgroup
#define THREADS 128         // 4 waves (wave32)
#define NT      16          // N tiles over H=256
#define HP      272         // h LDS pitch (halves), padded
#define CP      528         // ctx LDS pitch (halves), padded

__device__ __forceinline__ v16h frag_cat(v8h lo, v8h hi) {
  v16h r;
  #pragma unroll
  for (int q = 0; q < 8; ++q) { r[q] = lo[q]; r[q + 8] = hi[q]; }
  return r;
}

__device__ __forceinline__ v8f wmma16(v16h a, v16h b, v8f c) {
  return __builtin_amdgcn_wmma_f32_16x16x32_f16(false, a, false, b, (short)0, c,
                                                false, false);
}

// ---------------------------------------------------------------------------
// Prep kernels: pre-swizzle weights into exact per-lane WMMA B-fragment order.
// Fragment layout per (kt,nt): lane L holds column n = nt*16+(L&15); its 16
// halves cover k = kt*32 + 8*(L>=16) + {0..7, 16..23}.
// ---------------------------------------------------------------------------
__global__ void argmm_pack_bfrag(const float* __restrict__ src,
                                 h16* __restrict__ dst, int K, int rowOff,
                                 int stride) {
  int f = blockIdx.x * 256 + threadIdx.x;
  if (f >= K * 256) return;
  int q    = f & 15;
  int lane = (f >> 4) & 31;
  int rest = f >> 9;
  int nt   = rest & 15;
  int kt   = rest >> 4;
  int n = nt * 16 + (lane & 15);
  int k = kt * 32 + ((lane >> 4) & 1) * 8 + (q & 7) + ((q >= 8) ? 16 : 0);
  dst[f] = (h16)src[(size_t)(rowOff + k) * stride + n];
}

// Wh (H,3KD) -> per-step head fragments [i][kt=8][nt=2][lane][16], cols 30,31 = 0
__global__ void argmm_pack_wh(const float* __restrict__ Wh,
                              h16* __restrict__ dst) {
  int f = blockIdx.x * 256 + threadIdx.x;  // 64*8*2*32*16 = 524288
  if (f >= 64 * 8 * 2 * 32 * 16) return;
  int q    = f & 15;
  int lane = (f >> 4) & 31;
  int rest = f >> 9;
  int nt   = rest & 1;  rest >>= 1;
  int kt   = rest & 7;
  int i    = rest >> 3;
  int c = nt * 16 + (lane & 15);
  int k = kt * 32 + ((lane >> 4) & 1) * 8 + (q & 7) + ((q >= 8) ? 16 : 0);
  float v = (c < 30) ? Wh[(size_t)k * (3 * KK * DD) + i * (3 * KK) + c] : 0.f;
  dst[f] = (h16)v;
}

// masktab[i][h] = b0[h] + sum_{j<i} W0[D+j][h]
__global__ void argmm_build_mtab(const float* __restrict__ W0,
                                 const float* __restrict__ b0,
                                 float* __restrict__ mtab) {
  int h = threadIdx.x;  // 256 threads
  float run = b0[h];
  for (int i = 0; i < DD; ++i) {
    mtab[i * HH + h] = run;
    run += W0[(size_t)(DD + i) * HH + h];
  }
}

// ---------------------------------------------------------------------------
// Main kernel
// ---------------------------------------------------------------------------
__global__ __launch_bounds__(THREADS) void argmm_main(
    const float* __restrict__ g_value, const float* __restrict__ g_ctx,
    const float* __restrict__ g_W0, const float* __restrict__ g_bb1,
    const float* __restrict__ g_bb2, const float* __restrict__ g_bh,
    const h16* __restrict__ wsW0c, const h16* __restrict__ wsWb1,
    const h16* __restrict__ wsWb2, const h16* __restrict__ wsWh,
    const float* __restrict__ g_mtab, float* __restrict__ g_out) {
  __shared__ __align__(16) char smem[124928];
  float* value_lds = (float*)smem;                 // 64*64*4   = 16384 B
  float* p_lds     = (float*)(smem + 16384);       // 64*32*4   =  8192 B
  h16*   hbuf      = (h16*)(smem + 24576);         // 64*272*2  = 34816 B
  h16*   wstage    = (h16*)(smem + 59392);         // 4kt stage = 65536 B
  h16*   ctxA      = (h16*)(smem + 24576);         // aliases hbuf+wstage

  const int t    = threadIdx.x;
  const int lane = t & 31;
  const int wave = t >> 5;
  const int m0   = wave * 16;
  const int hi8  = ((lane >> 4) & 1) * 8;
  const int ln   = lane & 15;
  const int row0 = blockIdx.x * ROWS;

  // --- stage this WG's value tile (f32) ---
  for (int e = t; e < ROWS * DD; e += THREADS)
    value_lds[e] = g_value[(size_t)(row0 + (e >> 6)) * DD + (e & 63)];

  // --- stage context tile as f16 A operand ---
  for (int e = t; e < ROWS * CC; e += THREADS) {
    int r = e >> 9, c = e & 511;
    ctxA[r * CP + c] = (h16)g_ctx[(size_t)(row0 + r) * CC + c];
  }
  __syncthreads();

  v8f zero8;
  #pragma unroll
  for (int j = 0; j < 8; ++j) zero8[j] = 0.f;

  v8f base[NT];
  #pragma unroll
  for (int n = 0; n < NT; ++n) base[n] = zero8;

  // --- base = context @ W0[128:,:]  (K = 512, B-frags direct from L2) ---
  for (int kt = 0; kt < CC / 32; ++kt) {
    const h16* ap = &ctxA[(m0 + ln) * CP + kt * 32 + hi8];
    v16h a = frag_cat(*(const v8h*)ap, *(const v8h*)(ap + 16));
    #pragma unroll
    for (int n = 0; n < NT; ++n) {
      const h16* bp = &wsW0c[((size_t)(kt * NT + n) * 32 + lane) * 16];
      v16h b = frag_cat(*(const v8h*)bp, *(const v8h*)(bp + 8));
      base[n] = wmma16(a, b, base[n]);
    }
  }
  __syncthreads();  // done with ctxA alias region

  float acc_lp = 0.f;

  for (int i = 0; i < DD; ++i) {
    // h = relu(base + masktab[i])   (masktab includes b0)
    v8f hreg[NT];
    #pragma unroll
    for (int n = 0; n < NT; ++n) {
      float mt = g_mtab[i * HH + n * 16 + ln];
      #pragma unroll
      for (int j = 0; j < 8; ++j) {
        float x = base[n][j] + mt;
        hreg[n][j] = x > 0.f ? x : 0.f;
      }
    }

    for (int r = 0; r < 2; ++r) {
      // write current h (f16) into LDS as A operand
      #pragma unroll
      for (int n = 0; n < NT; ++n)
        #pragma unroll
        for (int j = 0; j < 8; ++j)
          hbuf[(m0 + j + hi8) * HP + n * 16 + ln] = (h16)hreg[n][j];
      __syncthreads();

      // ---- tmp = relu(h @ Wb1[r] + bb1[r]) ----
      v8f acc[NT];
      #pragma unroll
      for (int n = 0; n < NT; ++n) acc[n] = zero8;

      for (int ph = 0; ph < 2; ++ph) {  // two 64KB K-phases
        const v8h* src = (const v8h*)wsWb1 + (size_t)r * 8192 + ph * 4096;
        v8h* dst = (v8h*)wstage;
        for (int e = t; e < 4096; e += THREADS) dst[e] = src[e];
        __syncthreads();
        for (int kl = 0; kl < 4; ++kl) {
          const h16* ap = &hbuf[(m0 + ln) * HP + (ph * 4 + kl) * 32 + hi8];
          v16h a = frag_cat(*(const v8h*)ap, *(const v8h*)(ap + 16));
          #pragma unroll
          for (int n = 0; n < NT; ++n) {
            const h16* bp = &wstage[((kl * NT + n) * 32 + lane) * 16];
            v16h b = frag_cat(*(const v8h*)bp, *(const v8h*)(bp + 8));
            acc[n] = wmma16(a, b, acc[n]);
          }
        }
        __syncthreads();
      }

      // bias + relu -> tmp into hbuf (replaces h; all reads done)
      #pragma unroll
      for (int n = 0; n < NT; ++n) {
        float b1 = g_bb1[r * HH + n * 16 + ln];
        #pragma unroll
        for (int j = 0; j < 8; ++j) {
          float x = acc[n][j] + b1;
          x = x > 0.f ? x : 0.f;
          hbuf[(m0 + j + hi8) * HP + n * 16 + ln] = (h16)x;
        }
      }
      __syncthreads();

      // ---- h = h + tmp @ Wb2[r] + bb2[r]  (accumulate onto h directly) ----
      #pragma unroll
      for (int n = 0; n < NT; ++n) acc[n] = hreg[n];

      for (int ph = 0; ph < 2; ++ph) {
        const v8h* src = (const v8h*)wsWb2 + (size_t)r * 8192 + ph * 4096;
        v8h* dst = (v8h*)wstage;
        for (int e = t; e < 4096; e += THREADS) dst[e] = src[e];
        __syncthreads();
        for (int kl = 0; kl < 4; ++kl) {
          const h16* ap = &hbuf[(m0 + ln) * HP + (ph * 4 + kl) * 32 + hi8];
          v16h a = frag_cat(*(const v8h*)ap, *(const v8h*)(ap + 16));
          #pragma unroll
          for (int n = 0; n < NT; ++n) {
            const h16* bp = &wstage[((kl * NT + n) * 32 + lane) * 16];
            v16h b = frag_cat(*(const v8h*)bp, *(const v8h*)(bp + 8));
            acc[n] = wmma16(a, b, acc[n]);
          }
        }
        __syncthreads();
      }
      #pragma unroll
      for (int n = 0; n < NT; ++n) {
        float b2 = g_bb2[r * HH + n * 16 + ln];
        #pragma unroll
        for (int j = 0; j < 8; ++j) hreg[n][j] = acc[n][j] + b2;
      }
    }  // r

    // ---- head: p = h @ Wh_r[:,i,:] + bh_r[i]  (N padded to 32) ----
    #pragma unroll
    for (int n = 0; n < NT; ++n)
      #pragma unroll
      for (int j = 0; j < 8; ++j)
        hbuf[(m0 + j + hi8) * HP + n * 16 + ln] = (h16)hreg[n][j];
    __syncthreads();

    v8f pacc[2];
    pacc[0] = zero8; pacc[1] = zero8;
    for (int kt = 0; kt < 8; ++kt) {
      const h16* ap = &hbuf[(m0 + ln) * HP + kt * 32 + hi8];
      v16h a = frag_cat(*(const v8h*)ap, *(const v8h*)(ap + 16));
      #pragma unroll
      for (int nt2 = 0; nt2 < 2; ++nt2) {
        const h16* bp = &wsWh[((((size_t)i * 8 + kt) * 2 + nt2) * 32 + lane) * 16];
        v16h b = frag_cat(*(const v8h*)bp, *(const v8h*)(bp + 8));
        pacc[nt2] = wmma16(a, b, pacc[nt2]);
      }
    }
    #pragma unroll
    for (int nt2 = 0; nt2 < 2; ++nt2) {
      int c = nt2 * 16 + ln;
      float bi = (c < 30) ? g_bh[i * 30 + c] : 0.f;
      #pragma unroll
      for (int j = 0; j < 8; ++j)
        p_lds[(m0 + j + hi8) * 32 + c] = pacc[nt2][j] + bi;
    }
    __syncthreads();

    // ---- per-row GMM log-likelihood (one lane per row) ----
    if (t < ROWS) {
      const float* pr = &p_lds[t * 32];
      float lg[KK];
      float m1 = -1e30f;
      #pragma unroll
      for (int k = 0; k < KK; ++k) { lg[k] = pr[k]; m1 = fmaxf(m1, lg[k]); }
      float se = 0.f;
      #pragma unroll
      for (int k = 0; k < KK; ++k) se += __expf(lg[k] - m1);
      float lse = m1 + __logf(se);
      float v = value_lds[t * DD + i];
      float g[KK];
      float m2 = -1e30f;
      #pragma unroll
      for (int k = 0; k < KK; ++k) {
        float mu = pr[KK + k];
        float x  = pr[2 * KK + k];
        float sp = (x > 20.f) ? x : log1pf(__expf(x));  // softplus
        float s  = sp + 1e-5f;
        float z  = (v - mu) / s;
        g[k] = lg[k] - 0.5f * z * z - __logf(s) - 0.9189385332046727f;
        m2 = fmaxf(m2, g[k]);
      }
      float se2 = 0.f;
      #pragma unroll
      for (int k = 0; k < KK; ++k) se2 += __expf(g[k] - m2);
      acc_lp += (m2 + __logf(se2)) - lse;
    }

    // ---- base += value[:, i] (x) W0[i, :]  (rank-1, in accumulator layout) ----
    float vv[8];
    #pragma unroll
    for (int j = 0; j < 8; ++j) vv[j] = value_lds[(m0 + j + hi8) * DD + i];
    #pragma unroll
    for (int n = 0; n < NT; ++n) {
      float w = g_W0[(size_t)i * HH + n * 16 + ln];
      #pragma unroll
      for (int j = 0; j < 8; ++j) base[n][j] += vv[j] * w;
    }
    __syncthreads();
  }  // i

  if (t < ROWS) g_out[row0 + t] = acc_lp;
}

// ---------------------------------------------------------------------------
// Launch
// ---------------------------------------------------------------------------
extern "C" void kernel_launch(void* const* d_in, const int* in_sizes, int n_in,
                              void* d_out, int out_size, void* d_ws,
                              size_t ws_size, hipStream_t stream) {
  const float* value = (const float*)d_in[0];
  const float* ctx   = (const float*)d_in[1];
  const float* W0    = (const float*)d_in[2];
  const float* b0    = (const float*)d_in[3];
  const float* Wb1   = (const float*)d_in[4];
  const float* bb1   = (const float*)d_in[5];
  const float* Wb2   = (const float*)d_in[6];
  const float* bb2   = (const float*)d_in[7];
  const float* Wh    = (const float*)d_in[8];
  const float* bh    = (const float*)d_in[9];
  float* out = (float*)d_out;

  char* ws = (char*)d_ws;
  h16*   wsW0c = (h16*)(ws + 0);        // 512*256 halves  = 262144 B
  h16*   wsWb1 = (h16*)(ws + 262144);   // 2*256*256 halves= 262144 B
  h16*   wsWb2 = (h16*)(ws + 524288);   // 262144 B
  h16*   wsWh  = (h16*)(ws + 786432);   // 64*8*2*32*16 halves = 1048576 B
  float* mtab  = (float*)(ws + 1310720);// 64*256*4 = 65536 B

  // prep: swizzle weights into WMMA B-fragment order, build mask table
  argmm_pack_bfrag<<<512, 256, 0, stream>>>(W0, wsW0c, CC, 2 * DD, HH);
  for (int r = 0; r < 2; ++r) {
    argmm_pack_bfrag<<<256, 256, 0, stream>>>(Wb1 + (size_t)r * HH * HH,
                                              wsWb1 + (size_t)r * HH * HH, HH,
                                              0, HH);
    argmm_pack_bfrag<<<256, 256, 0, stream>>>(Wb2 + (size_t)r * HH * HH,
                                              wsWb2 + (size_t)r * HH * HH, HH,
                                              0, HH);
  }
  argmm_pack_wh<<<1024, 256, 0, stream>>>(Wh, wsWh);
  argmm_build_mtab<<<1, 256, 0, stream>>>(W0, b0, mtab);

  argmm_main<<<NB / ROWS, THREADS, 0, stream>>>(value, ctx, W0, bb1, bb2, bh,
                                                wsW0c, wsWb1, wsWb2, wsWh,
                                                mtab, out);
}
